// GraphAttentionNetwork_74758200754438
// MI455X (gfx1250) — compile-verified, matrix operand order checked by reading, not run on hardware
//
#include <hip/hip_runtime.h>
#include <hip/hip_bf16.h>

typedef __attribute__((ext_vector_type(16))) __bf16 v16bf;
typedef __attribute__((ext_vector_type(8)))  float  v8f;

#define BM 64
#define BN 128
#define BK 32
#define LDA 40   // bf16 elems per LDS row (pad 32->40: 80B rows keep 16B chunk alignment)
#define LDB 40

// ---- CDNA5 async copy: 16B global -> LDS, tracked by ASYNCcnt ---------------
__device__ __forceinline__ void async_copy16(unsigned lds_addr, unsigned long long gaddr) {
    asm volatile("global_load_async_to_lds_b128 %0, %1, off"
                 :: "v"(lds_addr), "v"(gaddr) : "memory");
}
__device__ __forceinline__ void wait_async0() {
    asm volatile("s_wait_asynccnt 0x0" ::: "memory");
}

// Stage one 64x32 A tile (1 chunk/thread) + 32x128 B tile (2 chunks/thread),
// both already bf16 in global, row-major along K. Branch-free (clamped rows).
__device__ __forceinline__ void stage_tile_async(
    const __bf16* __restrict__ Abf, const __bf16* __restrict__ Wt,
    __bf16* As, __bf16* Bs, int tid, int m0, int n0, int k0, int M, int K)
{
    {   // A: 64 rows x 4 chunks of 8 bf16
        int r = tid >> 2, c = tid & 3;
        int gm = min(m0 + r, M - 1);
        unsigned l = (unsigned)(size_t)(As + r * LDA + c * 8);
        unsigned long long g = (unsigned long long)(size_t)(Abf + (size_t)gm * K + k0 + c * 8);
        async_copy16(l, g);
    }
    #pragma unroll
    for (int i = 0; i < 2; ++i) {  // B: 128 rows x 4 chunks of 8 bf16
        int e = tid + i * 256;
        int nn = e >> 2, c = e & 3;
        unsigned l = (unsigned)(size_t)(Bs + nn * LDB + c * 8);
        unsigned long long g = (unsigned long long)(size_t)(Wt + (size_t)(n0 + nn) * K + k0 + c * 8);
        async_copy16(l, g);
    }
}

// ---------------- GEMM: C[M,N] = A[M,K] @ B[K,N], bf16 WMMA, f32 accum -------
// A pre-converted to bf16 row-major, B pre-converted+transposed to Wt[N][K].
__global__ __launch_bounds__(256)
void gemm_bf16_wmma(const __bf16* __restrict__ Abf, const __bf16* __restrict__ Wt,
                    float* __restrict__ C, int M, int K, int N)
{
    __shared__ __align__(16) __bf16 As[2 * BM * LDA];   // double-buffered [64][40]
    __shared__ __align__(16) __bf16 Bs[2 * BN * LDB];   // double-buffered [128][40] = (N,K)

    const int tid  = threadIdx.x;
    const int lane = tid & 31;
    const int wave = tid >> 5;
    const int wm   = wave & 3;    // 4 waves along M
    const int wn   = wave >> 2;   // 2 waves along N
    const int m0   = blockIdx.x * BM;
    const int n0   = blockIdx.y * BN;

    v8f acc[4] = {};
    const int nk = K / BK;

    // prologue: tile 0 -> buffer 0
    stage_tile_async(Abf, Wt, As, Bs, tid, m0, n0, 0, M, K);

    for (int kt = 0; kt < nk; ++kt) {
        wait_async0();        // my async copies for tile kt landed in LDS
        __syncthreads();      // everyone's landed; prior readers of next buffer done

        if (kt + 1 < nk)      // overlap: tile kt+1 in flight while we do WMMAs
            stage_tile_async(Abf, Wt,
                             As + ((kt + 1) & 1) * (BM * LDA),
                             Bs + ((kt + 1) & 1) * (BN * LDB),
                             tid, m0, n0, (kt + 1) * BK, M, K);
        if (kt + 2 < nk) {    // speculative L2 prefetch two tiles ahead
            int pr = min(m0 + (tid >> 2), M - 1);
            __builtin_prefetch(Abf + (size_t)pr * K + (kt + 2) * BK, 0, 1);
        }

        const __bf16* AsC = As + (kt & 1) * (BM * LDA);
        const __bf16* BsC = Bs + (kt & 1) * (BN * LDB);

        // A fragment: lane L<16 -> row M=L, K {0..7,16..23}; L>=16 -> K {8..15,24..31}
        const int arow = wm * 16 + (lane & 15);
        const int ksel = (lane >> 4) * 8;
        union { v16bf v; uint4 q[2]; } ua;
        ua.q[0] = *(const uint4*)&AsC[arow * LDA + ksel];
        ua.q[1] = *(const uint4*)&AsC[arow * LDA + ksel + 16];

        #pragma unroll
        for (int t = 0; t < 4; ++t) {
            const int bcol = wn * 64 + t * 16 + (lane & 15);
            union { v16bf v; uint4 q[2]; } ub;
            ub.q[0] = *(const uint4*)&BsC[bcol * LDB + ksel];
            ub.q[1] = *(const uint4*)&BsC[bcol * LDB + ksel + 16];
            acc[t] = __builtin_amdgcn_wmma_f32_16x16x32_bf16(
                false, ua.v, false, ub.v, (short)0, acc[t], false, false);
        }
    }

    // C/D layout: VGPR r, lanes 0-15 -> M=r, lanes 16-31 -> M=r+8; N = lane&15
    #pragma unroll
    for (int t = 0; t < 4; ++t) {
        #pragma unroll
        for (int r = 0; r < 8; ++r) {
            int row = m0 + wm * 16 + r + (lane >> 4) * 8;
            int col = n0 + wn * 64 + t * 16 + (lane & 15);
            if (row < M) C[(long)row * N + col] = acc[t][r];
        }
    }
}

// ---------------- fp32 -> bf16 conversions -----------------------------------
__global__ __launch_bounds__(256)
void cvt_f32_bf16_kernel(const float* __restrict__ s, __bf16* __restrict__ d, long n)
{
    long i = (long)blockIdx.x * blockDim.x + threadIdx.x;
    if (i < n) d[i] = (__bf16)s[i];
}

__global__ __launch_bounds__(256)
void cvt_wt_kernel(const float* __restrict__ W, __bf16* __restrict__ Wt, int K, int N)
{
    int i = blockIdx.x * blockDim.x + threadIdx.x;
    if (i >= K * N) return;
    int k = i / N, n = i % N;
    Wt[(size_t)n * K + k] = (__bf16)W[i];   // [K][N] -> [N][K]
}

// ---------------- attention logits: al = einsum('nhc,hc->nh') ----------------
__global__ __launch_bounds__(256)
void attn_logits_kernel(const float* __restrict__ h, const float* __restrict__ asrc,
                        const float* __restrict__ adst, float* __restrict__ als,
                        float* __restrict__ ald, int Nn, int H, int C)
{
    int idx = blockIdx.x * blockDim.x + threadIdx.x;
    if (idx >= Nn * H) return;
    int n = idx / H, hh = idx % H;
    const float* hp = h + (long)n * H * C + (long)hh * C;
    const float* as = asrc + hh * C;
    const float* ad = adst + hh * C;
    float s0 = 0.f, s1 = 0.f;
    for (int c = 0; c < C; c += 4) {
        float4 hv = *(const float4*)(hp + c);
        float4 av = *(const float4*)(as + c);
        float4 dv = *(const float4*)(ad + c);
        s0 += hv.x * av.x + hv.y * av.y + hv.z * av.z + hv.w * av.w;
        s1 += hv.x * dv.x + hv.y * dv.y + hv.z * dv.z + hv.w * dv.w;
    }
    als[idx] = s0;
    ald[idx] = s1;
}

// monotone float->uint key so atomicMax(u32) orders floats (incl. negatives)
__device__ __forceinline__ unsigned f2ord(float f) {
    unsigned u = __float_as_uint(f);
    return (u & 0x80000000u) ? ~u : (u | 0x80000000u);
}
__device__ __forceinline__ float ord2f(unsigned u) {
    unsigned v = (u & 0x80000000u) ? (u & 0x7FFFFFFFu) : ~u;
    return __uint_as_float(v);
}

// ---------------- edge pass 1: leaky_relu + segment max ----------------------
__global__ __launch_bounds__(256)
void edge_max_kernel(const long long* __restrict__ ei, const float* __restrict__ als,
                     const float* __restrict__ ald, float* __restrict__ tbuf,
                     unsigned* __restrict__ emax, int E, int H)
{
    int idx = blockIdx.x * blockDim.x + threadIdx.x;
    if (idx >= E * H) return;
    int e = idx / H, hh = idx % H;
    int src = (int)ei[e];
    int dst = (int)ei[E + e];
    float t = als[src * H + hh] + ald[dst * H + hh];
    t = t > 0.f ? t : 0.2f * t;                 // leaky_relu, PyG default slope
    tbuf[idx] = t;
    atomicMax(&emax[dst * H + hh], f2ord(t));
}

// ---------------- edge pass 2: exp + segment sum -----------------------------
__global__ __launch_bounds__(256)
void edge_exp_kernel(const long long* __restrict__ ei, const float* __restrict__ tbuf,
                     const unsigned* __restrict__ emax, float* __restrict__ pbuf,
                     float* __restrict__ denom, int E, int H)
{
    int idx = blockIdx.x * blockDim.x + threadIdx.x;
    if (idx >= E * H) return;
    int e = idx / H, hh = idx % H;
    int dst = (int)ei[E + e];
    float m = ord2f(emax[dst * H + hh]);
    float p = __expf(tbuf[idx] - m);
    pbuf[idx] = p;
    atomicAdd(&denom[dst * H + hh], p);
}

// ---------------- edge pass 3: weighted scatter-aggregate (dominant) ---------
__global__ __launch_bounds__(256)
void edge_aggregate_kernel(const long long* __restrict__ ei, const float* __restrict__ h,
                           const float* __restrict__ pbuf, const float* __restrict__ denom,
                           float* __restrict__ out, int E, int H, int C)
{
    long total = (long)E * H * (C / 4);
    long idx = (long)blockIdx.x * blockDim.x + threadIdx.x;
    if (idx >= total) return;
    int perEdge = H * (C / 4);
    int e  = (int)(idx / perEdge);
    int r  = (int)(idx % perEdge);
    int hh = r / (C / 4);
    int c4 = (r % (C / 4)) * 4;
    int src = (int)ei[e];
    int dst = (int)ei[E + e];
    float alpha = pbuf[(long)e * H + hh] / (denom[dst * H + hh] + 1e-16f);
    float4 hv = *(const float4*)(h + (long)src * H * C + (long)hh * C + c4);
    float* op = out + (long)dst * H * C + (long)hh * C + c4;
    atomicAdd(op + 0, alpha * hv.x);
    atomicAdd(op + 1, alpha * hv.y);
    atomicAdd(op + 2, alpha * hv.z);
    atomicAdd(op + 3, alpha * hv.w);
}

// ---------------- bias (+ optional ELU) --------------------------------------
__global__ __launch_bounds__(256)
void bias_act_kernel(float* __restrict__ out, const float* __restrict__ b,
                     long total, int D, int apply_elu)
{
    long idx = (long)blockIdx.x * blockDim.x + threadIdx.x;
    if (idx >= total) return;
    float v = out[idx] + b[idx % D];
    if (apply_elu) v = v > 0.f ? v : (__expf(v) - 1.f);
    out[idx] = v;
}

extern "C" void kernel_launch(void* const* d_in, const int* in_sizes, int n_in,
                              void* d_out, int out_size, void* d_ws, size_t ws_size,
                              hipStream_t stream)
{
    const float*     x   = (const float*)d_in[0];
    const long long* ei  = (const long long*)d_in[1];   // int64 edge_index [2,E]
    const float* W1 = (const float*)d_in[2];
    const float* a1s = (const float*)d_in[3];
    const float* a1d = (const float*)d_in[4];
    const float* b1  = (const float*)d_in[5];
    const float* W2 = (const float*)d_in[6];
    const float* a2s = (const float*)d_in[7];
    const float* a2d = (const float*)d_in[8];
    const float* b2  = (const float*)d_in[9];
    const float* W3 = (const float*)d_in[10];
    const float* a3s = (const float*)d_in[11];
    const float* a3d = (const float*)d_in[12];
    const float* b3  = (const float*)d_in[13];

    const int E   = in_sizes[1] / 2;          // 160000
    const int D1  = in_sizes[5];              // 512 = H*Hd
    const int Fin = in_sizes[2] / D1;         // 128
    const int N   = in_sizes[0] / Fin;        // 20000
    const int Fout = in_sizes[13];            // 128
    const int H = 4, C = D1 / 4;              // 4 heads x 128

    // workspace carve-up
    float*    hA    = (float*)d_ws;                       // N*D1
    float*    hB    = hA + (size_t)N * D1;                // N*D1
    float*    als   = hB + (size_t)N * D1;                // N*H
    float*    ald   = als + (size_t)N * H;                // N*H
    unsigned* emax  = (unsigned*)(ald + (size_t)N * H);   // N*H
    float*    denom = (float*)(emax + (size_t)N * H);     // N*H
    float*    tbuf  = denom + (size_t)N * H;              // E*H
    float*    pbuf  = tbuf + (size_t)E * H;               // E*H
    __bf16*   Abf   = (__bf16*)(pbuf + (size_t)E * H);    // N*D1 bf16
    __bf16*   Wt    = Abf + (size_t)N * D1;               // D1*D1 bf16 (max)

    auto run_layer = [&](const float* feat, int Fin_, const float* Wm,
                         const float* as_, const float* ad_, const float* bb,
                         int Hh, int Cc, float* gemm_out, float* agg_out, int elu) {
        const int D = Hh * Cc;

        long na = (long)N * Fin_;
        cvt_f32_bf16_kernel<<<(unsigned)((na + 255) / 256), 256, 0, stream>>>(feat, Abf, na);
        int nw = Fin_ * D;
        cvt_wt_kernel<<<(nw + 255) / 256, 256, 0, stream>>>(Wm, Wt, Fin_, D);

        dim3 gg((N + BM - 1) / BM, D / BN);
        gemm_bf16_wmma<<<gg, dim3(256), 0, stream>>>(Abf, Wt, gemm_out, N, Fin_, D);

        const int nh = N * Hh;
        attn_logits_kernel<<<(nh + 255) / 256, 256, 0, stream>>>(
            gemm_out, as_, ad_, als, ald, N, Hh, Cc);

        hipMemsetAsync(emax, 0, (size_t)nh * 4, stream);   // key 0 < key(-inf)
        hipMemsetAsync(denom, 0, (size_t)nh * 4, stream);
        hipMemsetAsync(agg_out, 0, (size_t)N * D * 4, stream);

        const int eh = E * Hh;
        edge_max_kernel<<<(eh + 255) / 256, 256, 0, stream>>>(ei, als, ald, tbuf, emax, E, Hh);
        edge_exp_kernel<<<(eh + 255) / 256, 256, 0, stream>>>(ei, tbuf, emax, pbuf, denom, E, Hh);

        long tot = (long)E * Hh * (Cc / 4);
        edge_aggregate_kernel<<<(unsigned)((tot + 255) / 256), 256, 0, stream>>>(
            ei, gemm_out, pbuf, denom, agg_out, E, Hh, Cc);

        long nd = (long)N * D;
        bias_act_kernel<<<(unsigned)((nd + 255) / 256), 256, 0, stream>>>(agg_out, bb, nd, D, elu);
    };

    // Layer 1: x[N,Fin] -> hB[N,512], ELU
    run_layer(x,  Fin, W1, a1s, a1d, b1, H, C, hA, hB, 1);
    // Layer 2: hB -> hB (gemm into hA, aggregate back into hB), ELU
    run_layer(hB, D1,  W2, a2s, a2d, b2, H, C, hA, hB, 1);
    // Layer 3: hB -> d_out, single head, no ELU (mean over 1 head == identity)
    run_layer(hB, D1,  W3, a3s, a3d, b3, 1, Fout, hA, (float*)d_out, 0);
}
